// kNN_VC_15582141350060
// MI455X (gfx1250) — matrix operand order repaired and checked by reading, not compiled, not run on hardware
//
#include <hip/hip_runtime.h>
#include <hip/hip_bf16.h>

// ---------------------------------------------------------------------------
// Cosine kNN (k=4) for MI455X (gfx1250, wave32, WMMA, async-LDS).
//
// Roofline: sim GEMM = 2*4096*65536*1024 ~ 0.55 TFLOP. 128-row M-blocks with
// a shared 16-col B tile in LDS give 128 flop/B (~4 GB total L2 traffic from
// the 128 MB L2-resident f16 target copy); N is split 8 ways for occupancy
// (grid 32x8 = 256 WGPs); B is double-buffered via
// global_load_async_to_lds_b128 so WMMAs overlap the B stream.
//
// Round-2 fix: B double-buffer selected by integer offset from the single
// __shared__ base (not a pointer array) so InferAddressSpaces emits
// ds_load_b128 for B fragments instead of generic flat_load_b128.
// ---------------------------------------------------------------------------

typedef __attribute__((ext_vector_type(16))) _Float16 v16h;
typedef __attribute__((ext_vector_type(8)))  _Float16 v8h;
typedef __attribute__((ext_vector_type(8)))  float    v8f;

union HalfVec16 { v16h v; v8h h[2]; };

#define KDIM     1024
#define M_BLK    128          // rows per workgroup (8 waves x 16)
#define NSLICES  8            // N split across blockIdx.y
#define N_TILE   16           // columns staged per iteration

// Map float -> uint so unsigned compare == float compare (totally ordered).
__device__ __forceinline__ unsigned ordf(float f) {
  unsigned u = __float_as_uint(f);
  return (u & 0x80000000u) ? ~u : (u | 0x80000000u);
}

// Insert key into sorted (desc) top-4 {t0 >= t1 >= t2 >= t3}.
__device__ __forceinline__ void ins4(unsigned long long k,
                                     unsigned long long& t0, unsigned long long& t1,
                                     unsigned long long& t2, unsigned long long& t3) {
  if (k > t3) {
    if (k > t0)      { t3 = t2; t2 = t1; t1 = t0; t0 = k; }
    else if (k > t1) { t3 = t2; t2 = t1; t1 = k; }
    else if (k > t2) { t3 = t2; t2 = k; }
    else             { t3 = k; }
  }
}

// Async copy of one 16-byte vector: global -> LDS (CDNA5 async path, ASYNCcnt).
__device__ __forceinline__ void async_b128(const _Float16* g, _Float16* l) {
  unsigned lds_off = (unsigned)(unsigned long long)l;  // low 32b of generic = LDS offset
  asm volatile("global_load_async_to_lds_b128 %0, %1, off"
               :: "v"(lds_off), "v"(g) : "memory");
}
__device__ __forceinline__ void wait_async0() {
  asm volatile("s_wait_asynccnt 0x0" ::: "memory");
}

// ---------------------------------------------------------------------------
// Kernel 1: per-row L2 normalize (f32 in) -> f16 out.  One block per row.
// ---------------------------------------------------------------------------
__global__ void __launch_bounds__(256) rownorm_f16_kernel(
    const float* __restrict__ in, _Float16* __restrict__ out, int ncols) {
  const int row = blockIdx.x;
  const float* r = in + (size_t)row * ncols;
  _Float16*    o = out + (size_t)row * ncols;
  const int tid = threadIdx.x;

  float ss = 0.0f;
  for (int c = tid; c < ncols; c += 256) { float v = r[c]; ss += v * v; }

  __shared__ float red[256];
  red[tid] = ss;
  __syncthreads();
  #pragma unroll
  for (int s = 128; s > 0; s >>= 1) {
    if (tid < s) red[tid] += red[tid + s];
    __syncthreads();
  }
  const float inv = 1.0f / fmaxf(sqrtf(red[0]), 1e-12f);
  for (int c = tid; c < ncols; c += 256) o[c] = (_Float16)(r[c] * inv);
}

// ---------------------------------------------------------------------------
// Kernel 2: sim = Sn @ Tn^T over a 128-row M-block and one N-slice, with
// streaming per-lane top-4; per-slice top-4 keys written to topws.
//   grid = (Ns/128, NSLICES), block = 256 (8 waves); wave w owns rows
//   [m0+16w, m0+16w+16).  LDS: A 256KB resident + B double buffer 2x32KB.
//   WMMA C layout: lane L holds rows {0..7}+8*(L>=16) of its tile, col L&15.
// ---------------------------------------------------------------------------
__global__ void __launch_bounds__(256, 1) gemm_topk_kernel(
    const _Float16* __restrict__ S16,   // [Ns, K] normalized f16
    const _Float16* __restrict__ T16,   // [Nt, K] normalized f16 (L2 resident)
    unsigned long long* __restrict__ topws, // [Ns, NSLICES, 4] u64 keys
    int Nt) {
  extern __shared__ char smem[];
  _Float16* As   = (_Float16*)smem;                              // 128*K f16
  _Float16* Bbuf = (_Float16*)(smem + (size_t)M_BLK * KDIM * 2); // 2*16*K f16
  unsigned long long* cand = (unsigned long long*)smem;          // overlay on A

  const int tid   = threadIdx.x;
  const int wave  = tid >> 5;
  const int lane  = tid & 31;
  const int m0    = blockIdx.x * M_BLK;
  const int slice = blockIdx.y;
  const int nPer  = Nt / NSLICES;
  const int nBase = slice * nPer;
  const int nIter = nPer / N_TILE;

  // ---- async stage: whole A block (128xK contiguous) + first B tile -------
  {
    const _Float16* Ag = S16 + (size_t)m0 * KDIM;
    const int nvec = (M_BLK * KDIM) / 8;          // 16384 x 16B
    for (int v = tid; v < nvec; v += 256)
      async_b128(Ag + (size_t)v * 8, As + (size_t)v * 8);
    const _Float16* Bg = T16 + (size_t)nBase * KDIM;
    #pragma unroll
    for (int i = 0; i < (N_TILE * KDIM / 8) / 256; ++i) {  // 8 vec/thread
      const int v = tid + i * 256;
      async_b128(Bg + (size_t)v * 8, Bbuf + (size_t)v * 8);
    }
  }
  wait_async0();
  __syncthreads();

  const int colInTile = lane & 15;
  const int hi        = lane >> 4;      // half-wave select
  const int aoff      = hi * 8;         // A K-offset per ISA 16-bit A layout
  const unsigned bfrag = (unsigned)colInTile * KDIM + (unsigned)hi * 8;
  const _Float16* Arow = As + (size_t)((wave << 4) + colInTile) * KDIM;

  unsigned long long t0[8], t1[8], t2[8], t3[8];
  #pragma unroll
  for (int r = 0; r < 8; ++r) { t0[r] = t1[r] = t2[r] = t3[r] = 0ull; }

  unsigned cur = 0;   // 0 or 1; buffer selected by pure offset arithmetic
  for (int it = 0; it < nIter; ++it) {
    const int n0 = nBase + it * N_TILE;

    // Prefetch next B tile into the other buffer (async, overlapped).
    if (it + 1 < nIter) {
      const _Float16* Bg = T16 + (size_t)(n0 + N_TILE) * KDIM;
      _Float16* Bd = Bbuf + (cur ^ 1u) * (N_TILE * KDIM);
      #pragma unroll
      for (int i = 0; i < (N_TILE * KDIM / 8) / 256; ++i) {
        const int v = tid + i * 256;
        async_b128(Bg + (size_t)v * 8, Bd + (size_t)v * 8);
      }
    }

    // Compute this wave's 16x16 tile over full K, A and B from LDS.
    const _Float16* Bs = Bbuf + cur * (N_TILE * KDIM) + bfrag;
    v8f acc = {};
    for (int k0 = 0; k0 < KDIM; k0 += 32) {
      HalfVec16 ua, ub;
      ua.h[0] = *(const v8h*)&Arow[k0 + aoff];
      ua.h[1] = *(const v8h*)&Arow[k0 + 16 + aoff];
      ub.h[0] = *(const v8h*)&Bs[k0];
      ub.h[1] = *(const v8h*)&Bs[k0 + 16];
      acc = __builtin_amdgcn_wmma_f32_16x16x32_f16(
          false, ua.v, false, ub.v, (short)0, acc, false, false);
    }

    const unsigned long long colbits = (unsigned)(n0 + colInTile);
    #pragma unroll
    for (int r = 0; r < 8; ++r) {
      const unsigned long long key =
          ((unsigned long long)ordf(acc[r]) << 32) | colbits;
      ins4(key, t0[r], t1[r], t2[r], t3[r]);
    }

    wait_async0();       // next B tile resident
    __syncthreads();     // everyone done reading Bbuf[cur], staging complete
    cur ^= 1u;
  }

  // ---- merge: dump per-lane candidates (overlay A region), reduce ---------
  __syncthreads();
  #pragma unroll
  for (int r = 0; r < 8; ++r) {
    const int lr = (wave << 4) + hi * 8 + r;     // local row 0..127
    unsigned long long* c = &cand[(size_t)lr * 64 + colInTile * 4];
    c[0] = t0[r]; c[1] = t1[r]; c[2] = t2[r]; c[3] = t3[r];
  }
  __syncthreads();

  if (tid < M_BLK) {
    unsigned long long b0 = 0, b1 = 0, b2 = 0, b3 = 0;
    const unsigned long long* c = &cand[(size_t)tid * 64];
    #pragma unroll 4
    for (int i = 0; i < 64; ++i) ins4(c[i], b0, b1, b2, b3);
    unsigned long long* o = &topws[((size_t)(m0 + tid) * NSLICES + slice) * 4];
    o[0] = b0; o[1] = b1; o[2] = b2; o[3] = b3;
  }
}

// ---------------------------------------------------------------------------
// Kernel 3: per row merge NSLICES*4 candidate keys -> top-4 indices, then
// gather raw target rows and average.  One block per source row.
// ---------------------------------------------------------------------------
__global__ void __launch_bounds__(256) merge_gather_kernel(
    const unsigned long long* __restrict__ topws,
    const float* __restrict__ Traw, float* __restrict__ out, int K) {
  const int row = blockIdx.x;
  __shared__ int idx4[4];
  if (threadIdx.x == 0) {
    unsigned long long b0 = 0, b1 = 0, b2 = 0, b3 = 0;
    const unsigned long long* c = &topws[(size_t)row * NSLICES * 4];
    #pragma unroll
    for (int i = 0; i < NSLICES * 4; ++i) ins4(c[i], b0, b1, b2, b3);
    idx4[0] = (int)(b0 & 0xffffffffu);
    idx4[1] = (int)(b1 & 0xffffffffu);
    idx4[2] = (int)(b2 & 0xffffffffu);
    idx4[3] = (int)(b3 & 0xffffffffu);
  }
  __syncthreads();
  const int i0 = idx4[0], i1 = idx4[1], i2 = idx4[2], i3 = idx4[3];
  for (int c = threadIdx.x; c < K; c += 256) {
    float s = Traw[(size_t)i0 * K + c] + Traw[(size_t)i1 * K + c] +
              Traw[(size_t)i2 * K + c] + Traw[(size_t)i3 * K + c];
    out[(size_t)row * K + c] = 0.25f * s;
  }
}

// ---------------------------------------------------------------------------
// Launcher.  ws: [Sn f16: Ns*K][Tn f16: Nt*K][topws: Ns*NSLICES*4 u64]
//   = 8 MB + 128 MB + 1 MB for the reference shapes.
// ---------------------------------------------------------------------------
extern "C" void kernel_launch(void* const* d_in, const int* in_sizes, int n_in,
                              void* d_out, int out_size, void* d_ws, size_t ws_size,
                              hipStream_t stream) {
  const float* src = (const float*)d_in[0];
  const float* tgt = (const float*)d_in[1];
  const int K  = KDIM;
  const int Ns = in_sizes[0] / K;      // 4096
  const int Nt = in_sizes[1] / K;      // 65536

  _Float16* src16 = (_Float16*)d_ws;
  _Float16* tgt16 = src16 + (size_t)Ns * K;
  unsigned long long* topws = (unsigned long long*)(tgt16 + (size_t)Nt * K);

  rownorm_f16_kernel<<<Ns, 256, 0, stream>>>(src, src16, K);
  rownorm_f16_kernel<<<Nt, 256, 0, stream>>>(tgt, tgt16, K);

  const size_t shmem = (size_t)M_BLK * K * sizeof(_Float16)        // A: 256 KB
                     + (size_t)2 * N_TILE * K * sizeof(_Float16);  // B: 64 KB
  dim3 grid(Ns / M_BLK, NSLICES);
  gemm_topk_kernel<<<grid, 256, shmem, stream>>>(src16, tgt16, topws, Nt);

  merge_gather_kernel<<<Ns, 256, 0, stream>>>(topws, tgt, (float*)d_out, K);
}
